// MultiHeadGraphAttention_37649683317089
// MI455X (gfx1250) — compile-verified, matrix operand order checked by reading, not run on hardware
//
#include <hip/hip_runtime.h>

typedef __attribute__((ext_vector_type(16))) _Float16 v16h;
typedef __attribute__((ext_vector_type(8)))  _Float16 v8h;
typedef __attribute__((ext_vector_type(8)))  float    v8f;

#define B_   8
#define N_   1024
#define IND  256
#define OUTD 256
#define H_   8
#define DK   32

// D = A(16x32 f16) x B(32x16 f16) + C, f32 accumulate
__device__ __forceinline__ v8f wmma16(v16h a, v16h b, v8f c) {
  return __builtin_amdgcn_wmma_f32_16x16x32_f16(
      /*neg_a*/false, a, /*neg_b*/false, b,
      /*c_mod*/(short)0, c, /*reuse_a*/false, /*reuse_b*/false);
}

// Load one 16x32 f16 WMMA fragment (A-layout; B-layout is identical with
// "row" meaning the output column and data stored [n][k]).
// Lane L<16:  row = row0+L,    K offsets k0+{0..7}  and k0+{16..23}
// Lane L>=16: row = row0+L-16, K offsets k0+{8..15} and k0+{24..31}
__device__ __forceinline__ v16h load_frag(const _Float16* __restrict__ base,
                                          int stride, int row0, int k0) {
  int lane = threadIdx.x & 31;
  int hi   = lane >> 4;
  const _Float16* p = base + (size_t)(row0 + (lane & 15)) * stride + k0 + hi * 8;
  v8h lo = *(const v8h*)(p);
  v8h hb = *(const v8h*)(p + 16);
  return __builtin_shufflevector(lo, hb, 0,1,2,3,4,5,6,7,8,9,10,11,12,13,14,15);
}

// Issue one per-lane 16-byte async global->LDS copy (gfx1250, ASYNCcnt).
__device__ __forceinline__ void async_b128(const void* gptr, void* lds_generic) {
  unsigned int  l = (unsigned int)(uintptr_t)lds_generic;   // low 32 bits = LDS byte addr
  unsigned long long g = (unsigned long long)(uintptr_t)gptr;
  asm volatile("global_load_async_to_lds_b128 %0, %1, off"
               :: "v"(l), "v"(g) : "memory");
}

__device__ __forceinline__ void wait_asynccnt0() {
  asm volatile("s_wait_asynccnt 0x0" ::: "memory");
}

// ---------------- conversion kernels ----------------
__global__ void cvt_x(const float* __restrict__ X, _Float16* __restrict__ Xh, int n) {
  int i = blockIdx.x * 256 + threadIdx.x;
  if (i < n) Xh[i] = (_Float16)X[i];
}

// W:[256 k][256 n] (row-major) -> WT:[n][k] f16
__global__ void cvt_wT(const float* __restrict__ W, _Float16* __restrict__ WT) {
  int i = blockIdx.x * 256 + threadIdx.x;   // 65536 threads
  int n = i & 255, k = i >> 8;              // coalesced read of W
  WT[n * 256 + k] = (_Float16)W[i];
}

// ---------------- fused QKV projection ----------------
// grid: (B*N/16) blocks, 256 threads (8 waves). Wave w owns 6 of the 48
// output 16-col tiles (3 matrices x 16 tiles).
__global__ void qkv_gemm(const _Float16* __restrict__ Xh,
                         const _Float16* __restrict__ WqT,
                         const _Float16* __restrict__ WkT,
                         const _Float16* __restrict__ WvT,
                         const float* __restrict__ bq,
                         const float* __restrict__ bk,
                         const float* __restrict__ bv,
                         _Float16* __restrict__ Qh,
                         _Float16* __restrict__ Kh,
                         _Float16* __restrict__ Vt) {
  int row0 = blockIdx.x * 16;
  int wave = threadIdx.x >> 5;
  int lane = threadIdx.x & 31;
  int hi = lane >> 4, nn = lane & 15;

  v8f acc[6];
#pragma unroll
  for (int i = 0; i < 6; ++i) acc[i] = (v8f)0.0f;

  for (int ks = 0; ks < 8; ++ks) {
    int k0 = ks * 32;
    v16h a = load_frag(Xh, IND, row0, k0);
#pragma unroll
    for (int i = 0; i < 6; ++i) {
      int t = wave * 6 + i;
      int mat = t >> 4, nt = t & 15;
      const _Float16* wt = (mat == 0) ? WqT : ((mat == 1) ? WkT : WvT);
      v16h b = load_frag(wt, IND, nt * 16, k0);
      acc[i] = wmma16(a, b, acc[i]);
    }
  }

#pragma unroll
  for (int i = 0; i < 6; ++i) {
    int t = wave * 6 + i;
    int mat = t >> 4, nt = t & 15;
    int n = nt * 16 + nn;
    const float* bp = (mat == 0) ? bq : ((mat == 1) ? bk : bv);
    float bias = bp[n];
#pragma unroll
    for (int r = 0; r < 8; ++r) {
      float v = acc[i][r] + bias;
      int m = r + hi * 8;
      int grow = row0 + m;                    // flat b*N + token
      if (mat == 0) {
        Qh[(size_t)grow * OUTD + n] = (_Float16)v;
      } else if (mat == 1) {
        Kh[(size_t)grow * OUTD + n] = (_Float16)v;
      } else {
        int b   = grow >> 10;                 // / N_
        int tok = grow & (N_ - 1);
        int h   = n >> 5;                     // / DK
        int dl  = n & (DK - 1);
        Vt[((size_t)(b * H_ + h) * DK + dl) * N_ + tok] = (_Float16)v;
      }
    }
  }
}

// ---------------- flash attention ----------------
// grid: B * (N/16) blocks, 256 threads = 8 waves = 8 heads.
// A/D 16x32 tiles stream through double-buffered LDS via async-to-LDS loads,
// shared by all 8 heads (captures the 8x head reuse of A/D in LDS).
__global__ void attn_kernel(const _Float16* __restrict__ Qh,
                            const _Float16* __restrict__ Kh,
                            const _Float16* __restrict__ Vt,
                            const float* __restrict__ Dm,
                            const int* __restrict__ Am,
                            const float* __restrict__ dist_scale_p,
                            _Float16* __restrict__ Oh) {
  __shared__ __align__(16) float    sD[2][16 * 32];
  __shared__ __align__(16) int      sA[2][16 * 32];
  __shared__ __align__(16) _Float16 sP[8][16 * 32];   // per-wave P tile

  int blk = blockIdx.x;
  int b  = blk >> 6;           // N/16 = 64 q-tiles
  int qt = blk & 63;
  int wave = threadIdx.x >> 5;
  int lane = threadIdx.x & 31;
  int h = wave;
  int q0 = qt * 16;
  int hi = lane >> 4, nn = lane & 15;

  const float dist_scale = *dist_scale_p;
  const float inv_sqrt_dk = 0.17677669529663687f;   // 1/sqrt(32)

  const _Float16* Qb = Qh + (size_t)b * N_ * OUTD;
  const _Float16* Kb = Kh + (size_t)b * N_ * OUTD;
  const _Float16* Vb = Vt + (size_t)(b * H_ + h) * DK * N_;
  const float* Db = Dm + (size_t)b * N_ * N_;
  const int*   Ab = Am + (size_t)b * N_ * N_;

  // Async A/D tile fetch: 256 threads x 16B = one 16x32 f32 D tile (threads
  // 0..127, i.e. waves 0-3) + one 16x32 i32 A tile (threads 128..255).
  int tid = threadIdx.x;
  int tch = tid & 127;               // chunk id within a tile
  int trow = tch >> 3;               // 8 x 16B chunks per 32-elem row
  int tcol = (tch & 7) * 4;

  // prologue: tile 0 -> buffer 0
  if (tid < 128)
    async_b128(Db + (size_t)(q0 + trow) * N_ + tcol, &sD[0][tch * 4]);
  else
    async_b128(Ab + (size_t)(q0 + trow) * N_ + tcol, &sA[0][tch * 4]);

  v16h qa = load_frag(Qb, OUTD, q0, h * DK);

  v8f o0 = (v8f)0.0f, o1 = (v8f)0.0f;
  float rm[8], rs[8];
#pragma unroll
  for (int r = 0; r < 8; ++r) { rm[r] = -3.0e38f; rs[r] = 0.0f; }

  _Float16* myP = &sP[wave][0];

  for (int kt2 = 0; kt2 < N_ / 32; ++kt2) {
    int cur = kt2 & 1;
    int kc0 = kt2 * 32;

    wait_asynccnt0();    // my wave's async tile for `cur` landed in LDS
    __syncthreads();     // everyone's landed; prev reads of buf[1-cur] done

    if (kt2 + 1 < N_ / 32) {   // software pipeline: fetch next tile
      int kn = kc0 + 32;
      if (tid < 128)
        async_b128(Db + (size_t)(q0 + trow) * N_ + kn + tcol, &sD[1 - cur][tch * 4]);
      else
        async_b128(Ab + (size_t)(q0 + trow) * N_ + kn + tcol, &sA[1 - cur][tch * 4]);
    }

    v16h kb0 = load_frag(Kb, OUTD, kc0,      h * DK);
    v16h kb1 = load_frag(Kb, OUTD, kc0 + 16, h * DK);
    v8f s0 = wmma16(qa, kb0, (v8f)0.0f);
    v8f s1 = wmma16(qa, kb1, (v8f)0.0f);

#pragma unroll
    for (int r = 0; r < 8; ++r) {
      int m = r + hi * 8;
      int c0 = nn, c1 = 16 + nn;
      float d0 = sD[cur][m * 32 + c0], d1 = sD[cur][m * 32 + c1];
      int   a0 = sA[cur][m * 32 + c0], a1 = sA[cur][m * 32 + c1];
      float v0 = (a0 <= 0) ? -1e9f : (s0[r] * inv_sqrt_dk - dist_scale * d0);
      float v1 = (a1 <= 0) ? -1e9f : (s1[r] * inv_sqrt_dk - dist_scale * d1);

      float t = fmaxf(v0, v1);           // row max over this 32-key slab
      t = fmaxf(t, __shfl_xor(t, 1, 32));
      t = fmaxf(t, __shfl_xor(t, 2, 32));
      t = fmaxf(t, __shfl_xor(t, 4, 32));
      t = fmaxf(t, __shfl_xor(t, 8, 32));

      float nm = fmaxf(rm[r], t);
      float scale = __expf(rm[r] - nm);
      float p0 = __expf(v0 - nm);
      float p1 = __expf(v1 - nm);
      float ps = p0 + p1;
      ps += __shfl_xor(ps, 1, 32);
      ps += __shfl_xor(ps, 2, 32);
      ps += __shfl_xor(ps, 4, 32);
      ps += __shfl_xor(ps, 8, 32);
      rs[r] = rs[r] * scale + ps;
      rm[r] = nm;
      o0[r] *= scale;
      o1[r] *= scale;

      myP[m * 32 + c0] = (_Float16)p0;
      myP[m * 32 + c1] = (_Float16)p1;
    }

    // P (C-layout) -> A-fragment via per-wave LDS round trip (in-order per wave)
    v16h pa  = load_frag(myP, 32, 0, 0);
    v16h vb0 = load_frag(Vb, N_, 0,  kc0);
    v16h vb1 = load_frag(Vb, N_, 16, kc0);
    o0 = wmma16(pa, vb0, o0);
    o1 = wmma16(pa, vb1, o1);
  }

  _Float16* Orow = Oh + (size_t)(b * N_ + q0) * OUTD + h * DK;
#pragma unroll
  for (int r = 0; r < 8; ++r) {
    float inv = 1.0f / rs[r];
    int m = r + hi * 8;
    Orow[(size_t)m * OUTD + nn]      = (_Float16)(o0[r] * inv);
    Orow[(size_t)m * OUTD + 16 + nn] = (_Float16)(o1[r] * inv);
  }
}

// ---------------- output projection ----------------
// grid: (B*N/16) blocks, 256 threads (8 waves x 2 n-tiles).
__global__ void out_gemm(const _Float16* __restrict__ Oh,
                         const _Float16* __restrict__ WoT,
                         const float* __restrict__ bo,
                         float* __restrict__ out) {
  int row0 = blockIdx.x * 16;
  int wave = threadIdx.x >> 5;
  int lane = threadIdx.x & 31;
  int hi = lane >> 4, nn = lane & 15;

  v8f acc[2];
  acc[0] = (v8f)0.0f; acc[1] = (v8f)0.0f;

  for (int ks = 0; ks < 8; ++ks) {
    int k0 = ks * 32;
    v16h a = load_frag(Oh, OUTD, row0, k0);
#pragma unroll
    for (int i = 0; i < 2; ++i) {
      int nt = wave * 2 + i;
      v16h b = load_frag(WoT, OUTD, nt * 16, k0);
      acc[i] = wmma16(a, b, acc[i]);
    }
  }

#pragma unroll
  for (int i = 0; i < 2; ++i) {
    int nt = wave * 2 + i;
    int n = nt * 16 + nn;
    float bias = bo[n];
#pragma unroll
    for (int r = 0; r < 8; ++r) {
      int m = r + hi * 8;
      out[(size_t)(row0 + m) * OUTD + n] = acc[i][r] + bias;
    }
  }
}

// ---------------- launcher ----------------
extern "C" void kernel_launch(void* const* d_in, const int* in_sizes, int n_in,
                              void* d_out, int out_size, void* d_ws, size_t ws_size,
                              hipStream_t stream) {
  (void)in_sizes; (void)n_in; (void)out_size; (void)ws_size;

  const float* X  = (const float*)d_in[0];
  const int*   A  = (const int*)  d_in[1];
  const float* Dm = (const float*)d_in[2];
  const float* Wq = (const float*)d_in[3];
  const float* bq = (const float*)d_in[4];
  const float* Wk = (const float*)d_in[5];
  const float* bk = (const float*)d_in[6];
  const float* Wv = (const float*)d_in[7];
  const float* bv = (const float*)d_in[8];
  const float* Wo = (const float*)d_in[9];
  const float* bo = (const float*)d_in[10];
  const float* ds = (const float*)d_in[11];
  float* out = (float*)d_out;

  char* ws = (char*)d_ws;
  const size_t MB = 1024 * 1024;
  _Float16* Xh  = (_Float16*)(ws + 0 * MB);   // B*N*256 f16 = 4MB
  _Float16* Qh  = (_Float16*)(ws + 4 * MB);
  _Float16* Kh  = (_Float16*)(ws + 8 * MB);
  _Float16* Vt  = (_Float16*)(ws + 12 * MB);  // [B][H][DK][N]
  _Float16* Oh  = (_Float16*)(ws + 16 * MB);
  _Float16* WqT = (_Float16*)(ws + 20 * MB);
  _Float16* WkT = (_Float16*)(ws + 20 * MB + 128 * 1024);
  _Float16* WvT = (_Float16*)(ws + 20 * MB + 256 * 1024);
  _Float16* WoT = (_Float16*)(ws + 20 * MB + 384 * 1024);

  const int nX = B_ * N_ * IND;                 // 2,097,152
  cvt_x<<<(nX + 255) / 256, 256, 0, stream>>>(X, Xh, nX);
  cvt_wT<<<256, 256, 0, stream>>>(Wq, WqT);
  cvt_wT<<<256, 256, 0, stream>>>(Wk, WkT);
  cvt_wT<<<256, 256, 0, stream>>>(Wv, WvT);
  cvt_wT<<<256, 256, 0, stream>>>(Wo, WoT);

  qkv_gemm<<<B_ * N_ / 16, 256, 0, stream>>>(Xh, WqT, WkT, WvT, bq, bk, bv,
                                             Qh, Kh, Vt);
  attn_kernel<<<B_ * (N_ / 16), 256, 0, stream>>>(Qh, Kh, Vt, Dm, A, ds, Oh);
  out_gemm<<<B_ * N_ / 16, 256, 0, stream>>>(Oh, WoT, bo, out);
}
